// MiniSTU_76493367542185
// MI455X (gfx1250) — matrix operand order repaired and checked by reading, not compiled, not run on hardware
//
#include <hip/hip_runtime.h>

// ---------------------------------------------------------------------------
// MiniSTU on MI455X (gfx1250): time-domain causal Toeplitz conv + projection,
// all matmuls via v_wmma_f32_16x16x32_bf16 (fp32 accumulate), U-tile staged in
// 197KB of WGP LDS. See analysis above for the roofline reasoning.
// ---------------------------------------------------------------------------

#define L_SEQ 2048
#define NK    24
#define NI    256
#define NO    256
#define NB    2
#define KI    (NK * NI)        // 6144 combined (k,i) contraction dim
#define U_STR (KI + 8)         // padded LDS row stride (halves): 6152 -> no bank conflicts
#define PR_LEN 2112            // reversed phi: 2048 valid + 64 zero pad (negative lags)
#define XT_N  (NB * NI * L_SEQ)    // 1,048,576 halves
#define PR_N  (2 * NK * PR_LEN)    //   101,376 halves
#define M2_N  (2 * NO * KI)        // 3,145,728 halves  (total ws ~ 8.6 MB)

typedef __attribute__((ext_vector_type(16))) __bf16 v16bf;
typedef __attribute__((ext_vector_type(8)))  float  v8f;
typedef __attribute__((ext_vector_type(4)))  unsigned int v4u;

union V16 {
  v16bf v;
  unsigned short u[16];
  v4u q[2];
};

__device__ __forceinline__ unsigned short f2bf(float f) {
  // round-to-nearest-even fp32 -> bf16
  unsigned int x = __float_as_uint(f);
  unsigned int r = x + 0x7FFFu + ((x >> 16) & 1u);
  return (unsigned short)(r >> 16);
}

// ---------------------------------------------------------------------------
// Prep: build bf16 operands in workspace.
//   xT[b][i][t]          : x transposed (B-operand of conv: contiguous in t)
//   phiRev[s][k][p]      : phi_s[2047-p], p>=2048 -> 0   (Toeplitz A gathers)
//   M2[s][o][ki]         : M_s[k,i,o] transposed (B-operand of projection)
// ---------------------------------------------------------------------------
__global__ void stu_prep(const float* __restrict__ x,
                         const float* __restrict__ phi,
                         const float* __restrict__ Mp,
                         const float* __restrict__ Mm,
                         unsigned short* __restrict__ ws) {
  unsigned short* xT = ws;
  unsigned short* pr = ws + XT_N;
  unsigned short* m2 = ws + XT_N + PR_N;
  const int stride = gridDim.x * blockDim.x;
  const int tid = blockIdx.x * blockDim.x + threadIdx.x;

  for (int idx = tid; idx < XT_N; idx += stride) {
    int b = idx / (NI * L_SEQ);
    int r = idx - b * (NI * L_SEQ);
    int i = r / L_SEQ;
    int t = r - i * L_SEQ;
    xT[idx] = f2bf(x[((size_t)b * L_SEQ + t) * NI + i]);
  }
  for (int idx = tid; idx < PR_N; idx += stride) {
    int s = idx / (NK * PR_LEN);
    int r = idx - s * (NK * PR_LEN);
    int k = r / PR_LEN;
    int p = r - k * PR_LEN;
    float v = 0.f;
    if (p < L_SEQ) {
      int m = (L_SEQ - 1) - p;
      v = phi[m * NK + k];
      if (s == 1 && (m & 1)) v = -v;   // phi_minus[m] = (-1)^m * phi[m]
    }
    pr[idx] = f2bf(v);
  }
  for (int idx = tid; idx < M2_N; idx += stride) {
    int s = idx / (NO * KI);
    int r = idx - s * (NO * KI);
    int o = r / KI;
    int ki = r - o * KI;                       // ki = k*256 + i
    const float* M = s ? Mm : Mp;              // M[(k*256+i)*256 + o]
    m2[idx] = f2bf(M[(size_t)ki * NO + o]);
  }
}

// ---------------------------------------------------------------------------
// Main fused kernel: one block per (batch, 16-row l-tile); 8 wave32 waves.
// Phase 1: U_s[16 x 6144] = Toeplitz(phi_s) x  -> bf16 in LDS (197KB)
// Phase 2: acc += U_s x M2_s                   -> fp32 accumulators
// Both signs accumulate into the same registers; epilogue stores fp32 out.
// ---------------------------------------------------------------------------
__global__ __launch_bounds__(256, 1)
void stu_main(const unsigned short* __restrict__ ws, float* __restrict__ out) {
  extern __shared__ unsigned short U[];   // 16 * U_STR halves = 196,864 bytes
  const unsigned short* xT     = ws;
  const unsigned short* phiRev = ws + XT_N;
  const unsigned short* M2     = ws + XT_N + PR_N;

  const int blk    = blockIdx.x;
  const int b      = blk >> 7;             // 128 l-tiles per batch
  const int l_base = (blk & 127) << 4;
  const int tid    = threadIdx.x;
  const int wave   = tid >> 5;
  const int lane   = tid & 31;
  const int row    = lane & 15;
  const int half   = lane >> 4;
  const int nt     = ((l_base + 15) >> 5) + 1;   // causal: only t <= l chunks

  const v8f vzero = {0.f, 0.f, 0.f, 0.f, 0.f, 0.f, 0.f, 0.f};
  v8f acc0 = vzero, acc1 = vzero;          // persist across both signs

  for (int s = 0; s < 2; ++s) {
    // ---------------- Phase 1: causal conv via Toeplitz WMMA ----------------
    for (int kk = 0; kk < 3; ++kk) {
      const int k = wave * 3 + kk;                         // 8 waves x 3 = 24
      const unsigned short* pk = phiRev + (size_t)(s * NK + k) * PR_LEN;
      v8f c[16];
      #pragma unroll
      for (int it = 0; it < 16; ++it) c[it] = vzero;

      for (int tc = 0; tc < nt; ++tc) {
        const int t_base = tc << 5;
        // A[l_local, t_local] = phi_s[(l_base+row) - (t_base+K)]; reversed phi
        // makes each lane's two 8-element K-runs contiguous & zero-padded.
        const int q0 = (L_SEQ - 1) - (l_base + row - t_base) + half * 8;
        V16 A;
        #pragma unroll
        for (int j = 0; j < 8; ++j) {
          A.u[j]     = pk[q0 + j];        // K = {0..7} / {8..15}
          A.u[8 + j] = pk[q0 + 16 + j];   // K = {16..23} / {24..31}
        }
        const int toff = t_base + half * 8;
        const unsigned short* xb =
            xT + (size_t)b * NI * L_SEQ + (size_t)row * L_SEQ + toff;
        #pragma unroll
        for (int it = 0; it < 16; ++it) {
          const unsigned short* bp = xb + (size_t)(it * 16) * L_SEQ;
          V16 Bv;
          Bv.q[0] = *(const v4u*)(bp);        // two 16B runs (K 8h.., 16+8h..), 16B aligned
          Bv.q[1] = *(const v4u*)(bp + 16);
          c[it] = __builtin_amdgcn_wmma_f32_16x16x32_bf16(
              false, A.v, false, Bv.v, (short)0, c[it], false, false);
        }
      }
      // spill the 16x256 slab for this k into the LDS U tile (bf16)
      #pragma unroll
      for (int it = 0; it < 16; ++it) {
        const int col = k * NI + it * 16 + row;
        #pragma unroll
        for (int r = 0; r < 8; ++r) {
          U[(r + half * 8) * U_STR + col] = f2bf(c[it][r]);
        }
      }
    }
    __syncthreads();

    // ---------------- Phase 2: projection U x M2_s ----------------
    const int ob = wave * 32;               // 8 waves cover all 256 outputs
    const unsigned short* m2s = M2 + (size_t)s * NO * KI;
    for (int kc = 0; kc < KI / 32; ++kc) {
      const int kib = kc << 5;
      V16 A;
      const unsigned short* up = U + row * U_STR + kib + half * 16;
      A.q[0] = *(const v4u*)(up);           // 16B aligned (padded row stride)
      A.q[1] = *(const v4u*)(up + 8);

      const unsigned short* bp0 =
          m2s + (size_t)(ob + row) * KI + kib + half * 16;
      V16 B0;
      B0.q[0] = *(const v4u*)(bp0);
      B0.q[1] = *(const v4u*)(bp0 + 8);
      acc0 = __builtin_amdgcn_wmma_f32_16x16x32_bf16(
          false, A.v, false, B0.v, (short)0, acc0, false, false);

      const unsigned short* bp1 = bp0 + (size_t)16 * KI;
      V16 B1;
      B1.q[0] = *(const v4u*)(bp1);
      B1.q[1] = *(const v4u*)(bp1 + 8);
      acc1 = __builtin_amdgcn_wmma_f32_16x16x32_bf16(
          false, A.v, false, B1.v, (short)0, acc1, false, false);
    }
    __syncthreads();   // before next sign overwrites U
  }

  // ---------------- Epilogue: fp32 store (C/D layout: M=r+half*8, N=lane&15)
  const size_t obase =
      ((size_t)(b << 11) + l_base) * NO + wave * 32 + row;
  #pragma unroll
  for (int r = 0; r < 8; ++r) {
    const size_t rowoff = (size_t)(r + half * 8) * NO;
    out[obase + rowoff]      = acc0[r];
    out[obase + rowoff + 16] = acc1[r];
  }
}

// ---------------------------------------------------------------------------
extern "C" void kernel_launch(void* const* d_in, const int* in_sizes, int n_in,
                              void* d_out, int out_size, void* d_ws, size_t ws_size,
                              hipStream_t stream) {
  const float* x   = (const float*)d_in[0];   // (B, L, I)
  const float* phi = (const float*)d_in[1];   // (L, K)
  const float* Mp  = (const float*)d_in[2];   // (K, I, O)
  const float* Mm  = (const float*)d_in[3];   // (K, I, O)
  unsigned short* ws = (unsigned short*)d_ws; // needs ~8.6 MB
  float* out = (float*)d_out;                 // (B, L, O) fp32

  stu_prep<<<1024, 256, 0, stream>>>(x, phi, Mp, Mm, ws);
  stu_main<<<NB * (L_SEQ / 16), 256, 16 * U_STR * sizeof(unsigned short), stream>>>(ws, out);
}
